// PointTransformerLayer_20194936226527
// MI455X (gfx1250) — matrix operand order repaired and checked by reading, not compiled, not run on hardware
//
#include <hip/hip_runtime.h>
#include <math.h>

#define Bz  16
#define Np  512
#define Cc  512
#define NSn 16
#define RSQ 0.9999950000374997f   // 1/sqrt(1+1e-5)

typedef __attribute__((ext_vector_type(16))) __bf16       v16bf;
typedef __attribute__((ext_vector_type(8)))  float        v8f;
typedef __attribute__((ext_vector_type(4)))  unsigned int u32x4;
typedef __attribute__((ext_vector_type(4)))  float        f32x4;

union FragU { v16bf v; u32x4 q[2]; };

__device__ __forceinline__ unsigned short f2bf(float f) {
  union { __bf16 h; unsigned short u; } cv; cv.h = (__bf16)f; return cv.u;
}

// A-fragment (16x32 bf16): lane half (l>>4) selects K interleave
// j=0..7  -> K = kb + 8*half + j
// j=8..15 -> K = kb + 16 + 8*half + (j-8)
__device__ __forceinline__ v16bf ld_fragA(const unsigned short* rowp, int kb, int half) {
  FragU f;
  const unsigned short* p0 = rowp + kb + 8 * half;
  f.q[0] = *reinterpret_cast<const u32x4*>(p0);
  f.q[1] = *reinterpret_cast<const u32x4*>(p0 + 16);
  return f.v;
}

// B-fragment (32x16 bf16, from transposed weights row n): same K interleave as A
__device__ __forceinline__ v16bf ld_fragB(const unsigned short* rowp, int kb, int half) {
  FragU f;
  const unsigned short* p0 = rowp + kb + 8 * half;
  f.q[0] = *reinterpret_cast<const u32x4*>(p0);
  f.q[1] = *reinterpret_cast<const u32x4*>(p0 + 16);
  return f.v;
}

// CDNA5 async global->LDS DMA: 16 bytes per lane, tracked by ASYNCcnt.
// GVS addressing: mem = SADDR(64b SGPR) + VADDR(32b byte offset); LDS addr from VDST.
__device__ __forceinline__ void async_ld16(unsigned ldsaddr, unsigned goff, const void* base) {
  asm volatile("global_load_async_to_lds_b128 %0, %1, %2"
               :: "v"(ldsaddr), "v"(goff), "s"(base)
               : "memory");
}
__device__ __forceinline__ void wait_async0() {
  asm volatile("s_wait_asynccnt 0" ::: "memory");
}

// ---------------- kNN: stable 16-smallest selection, reference distance formula ----
__global__ void knn_kernel(const float* __restrict__ p, int* __restrict__ idx) {
  __shared__ float pl[Np * 3];
  __shared__ float sql[Np];
  int tid = threadIdx.x;
  int gid = blockIdx.x * 256 + tid;
  int b = gid >> 9;
  int n = gid & (Np - 1);
  const float* pb = p + (size_t)b * Np * 3;
  for (int i = tid; i < Np * 3; i += 256) pl[i] = pb[i];
  __syncthreads();
  for (int m = tid; m < Np; m += 256) {
    float a0 = pl[m * 3], a1 = pl[m * 3 + 1], a2 = pl[m * 3 + 2];
    sql[m] = a0 * a0 + a1 * a1 + a2 * a2;
  }
  __syncthreads();
  float qx = pl[n * 3], qy = pl[n * 3 + 1], qz = pl[n * 3 + 2];
  float sqn = sql[n];
  float prev_d = -3.4e38f; int prev_i = -1;
  int* op = idx + (size_t)gid * NSn;
  for (int t = 0; t < NSn; t++) {
    float bd = 3.4e38f; int bi = Np;
    for (int m = 0; m < Np; m++) {
      float d = sqn + sql[m] -
                2.0f * (qx * pl[m * 3] + qy * pl[m * 3 + 1] + qz * pl[m * 3 + 2]);
      bool gt = (d > prev_d) || (d == prev_d && m > prev_i);
      bool lt = (d < bd) || (d == bd && m < bi);
      if (gt && lt) { bd = d; bi = m; }
    }
    op[t] = bi; prev_d = bd; prev_i = bi;
  }
}

// ---------------- transpose + f32 -> bf16 convert: out[c][r] = in[r][c] ----------
__global__ void tconv_kernel(const float* __restrict__ in, unsigned short* __restrict__ out,
                             int R, int C, long in_bs, long out_bs) {
  __shared__ float tile[32][33];
  in += (size_t)blockIdx.z * in_bs;
  out += (size_t)blockIdx.z * out_bs;
  int c0 = blockIdx.x * 32, r0 = blockIdx.y * 32;
  int tx = threadIdx.x, ty = threadIdx.y;
  for (int i = 0; i < 32; i += 8) {
    int r = r0 + ty + i, c = c0 + tx;
    if (r < R && c < C) tile[ty + i][tx] = in[(size_t)r * C + c];
  }
  __syncthreads();
  for (int i = 0; i < 32; i += 8) {
    int c = c0 + ty + i, r = r0 + tx;
    if (r < R && c < C) out[(size_t)c * R + r] = f2bf(tile[tx][ty + i]);
  }
}

// ---------------- QKV GEMM: C[8192x512] = A @ W + bias ---------------------------
// 128x128 block tile, K-step 32, double-buffered LDS fed by async global->LDS DMA.
// 8 waves; each wave owns a 32(M)x64(N) sub-tile = 2x4 WMMA tiles.
__global__ void __launch_bounds__(256)
gemm_kernel(const unsigned short* __restrict__ A, const unsigned short* __restrict__ Wt,
            const float* __restrict__ bias, float* __restrict__ out) {
  __shared__ __align__(16) unsigned short As[2][128][40]; // 32 cols + pad (80B pitch)
  __shared__ __align__(16) unsigned short Bs[2][128][40];

  int tid = threadIdx.x;
  int l = tid & 31, wv = tid >> 5;
  int row = l & 15, half = l >> 4;
  int wm = wv >> 1, wn = wv & 1;           // 4 x 2 wave grid
  int M0 = blockIdx.y * 128;
  int N0 = blockIdx.x * 128;

  // loader: each thread owns one tile row (tid>>1) and a 16-short column group
  int lrow = tid >> 1;
  int lcol = (tid & 1) * 16;                          // in shorts
  unsigned aoff = (unsigned)(((M0 + lrow) * Cc + lcol) * 2); // byte offsets
  unsigned boff = (unsigned)(((N0 + lrow) * Cc + lcol) * 2);
  unsigned lA = (unsigned)(size_t)&As[0][lrow][lcol];
  unsigned lB = (unsigned)(size_t)&Bs[0][lrow][lcol];
  unsigned bufA = (unsigned)((size_t)&As[1][0][0] - (size_t)&As[0][0][0]);
  unsigned bufB = (unsigned)((size_t)&Bs[1][0][0] - (size_t)&Bs[0][0][0]);

  // prefetch K-tile 0 into buffer 0
  async_ld16(lA,      aoff,      A);
  async_ld16(lA + 16, aoff + 16, A);
  async_ld16(lB,      boff,      Wt);
  async_ld16(lB + 16, boff + 16, Wt);
  wait_async0();
  __syncthreads();

  v8f acc[2][4] = {};
  for (int kb = 0; kb < Cc; kb += 32) {
    int buf = (kb >> 5) & 1;
    int nbuf = buf ^ 1;
    if (kb + 32 < Cc) {  // kick off DMA of next K-tile while we compute
      unsigned ga = aoff + (unsigned)((kb + 32) * 2);
      unsigned gb = boff + (unsigned)((kb + 32) * 2);
      async_ld16(lA + nbuf * bufA,      ga,      A);
      async_ld16(lA + nbuf * bufA + 16, ga + 16, A);
      async_ld16(lB + nbuf * bufB,      gb,      Wt);
      async_ld16(lB + nbuf * bufB + 16, gb + 16, Wt);
    }
    v16bf a0 = ld_fragA(&As[buf][wm * 32 + row][0], 0, half);
    v16bf a1 = ld_fragA(&As[buf][wm * 32 + 16 + row][0], 0, half);
#pragma unroll
    for (int nt = 0; nt < 4; nt++) {
      v16bf bf = ld_fragB(&Bs[buf][wn * 64 + nt * 16 + row][0], 0, half);
      acc[0][nt] = __builtin_amdgcn_wmma_f32_16x16x32_bf16(false, a0, false, bf, (short)0, acc[0][nt], false, false);
      acc[1][nt] = __builtin_amdgcn_wmma_f32_16x16x32_bf16(false, a1, false, bf, (short)0, acc[1][nt], false, false);
    }
    wait_async0();     // next tile fully in LDS
    __syncthreads();   // all waves done reading current buffer
  }

#pragma unroll
  for (int mt = 0; mt < 2; mt++) {
#pragma unroll
    for (int nt = 0; nt < 4; nt++) {
      int n = N0 + wn * 64 + nt * 16 + row;
      float bc = bias[n];
#pragma unroll
      for (int i = 0; i < 8; i++) {
        int m = M0 + wm * 32 + mt * 16 + i + 8 * half;
        out[(size_t)m * Cc + n] = acc[mt][nt][i] + bc;
      }
    }
  }
}

// ---------------- fused attention: one wave per point ----------------------------
__global__ void __launch_bounds__(128)
attn_kernel(const float* __restrict__ p, const int* __restrict__ idx,
            const float* __restrict__ xq, const float* __restrict__ xk,
            const float* __restrict__ xv,
            const unsigned short* __restrict__ Wt1, const unsigned short* __restrict__ Wt2,
            const float* __restrict__ p1W, const float* __restrict__ p1b,
            const float* __restrict__ pbg, const float* __restrict__ pbb,
            const float* __restrict__ p2W, const float* __restrict__ p2b,
            const float* __restrict__ g1, const float* __restrict__ b1,
            const float* __restrict__ w1b,
            const float* __restrict__ g2, const float* __restrict__ b2,
            const float* __restrict__ w2b,
            float* __restrict__ out) {
  __shared__ __align__(16) unsigned short t2s[4][16][72]; // bf16, padded rows
  __shared__ float wts[4][16][64];
  __shared__ float hs[4][16][3];
  __shared__ int   nbs[4][16];

  int l = threadIdx.x & 31, wv = threadIdx.x >> 5;
  int s = l & 15, half = l >> 4;
  int gid = blockIdx.x * 4 + wv;          // b*512 + n
  int b = gid >> 9;

  int nb = idx[(size_t)gid * NSn + s];
  int nrow = b * Np + nb;
  const float* pc = p + (size_t)gid * 3;
  const float* pn = p + (size_t)nrow * 3;
  float rx = pn[0] - pc[0], ry = pn[1] - pc[1], rz = pn[2] - pc[2];
  float h[3];
#pragma unroll
  for (int d = 0; d < 3; d++) {
    float q = rx * p1W[0 * 3 + d] + ry * p1W[1 * 3 + d] + rz * p1W[2 * 3 + d] + p1b[d];
    float t = fmaf(q, pbg[d] * RSQ, pbb[d]);
    h[d] = t > 0.f ? t : 0.f;
  }
  nbs[wv][s] = nb;
  hs[wv][s][0] = h[0]; hs[wv][s][1] = h[1]; hs[wv][s][2] = h[2];

  const float* krow = xk + (size_t)nrow * Cc;
  const float* qrow = xq + (size_t)gid * Cc;

  // ---- w1 GEMM: u[16x64] = t1[16x512] @ Wt1^T, A built in-register in WMMA layout
  v8f acc[4] = {{}, {}, {}, {}};
  for (int kb = 0; kb < Cc; kb += 32) {
    FragU fa;
#pragma unroll
    for (int r2 = 0; r2 < 2; r2++) {
      int cb = kb + 16 * r2 + 8 * half;
      f32x4 kv0 = *(const f32x4*)(krow + cb),        kv1 = *(const f32x4*)(krow + cb + 4);
      f32x4 qv0 = *(const f32x4*)(qrow + cb),        qv1 = *(const f32x4*)(qrow + cb + 4);
      f32x4 wa0 = *(const f32x4*)(p2W + cb),         wa1 = *(const f32x4*)(p2W + cb + 4);
      f32x4 wb0 = *(const f32x4*)(p2W + Cc + cb),    wb1 = *(const f32x4*)(p2W + Cc + cb + 4);
      f32x4 wc0 = *(const f32x4*)(p2W + 2*Cc + cb),  wc1 = *(const f32x4*)(p2W + 2*Cc + cb + 4);
      f32x4 pb0 = *(const f32x4*)(p2b + cb),         pb1 = *(const f32x4*)(p2b + cb + 4);
      f32x4 gg0 = *(const f32x4*)(g1 + cb),          gg1 = *(const f32x4*)(g1 + cb + 4);
      f32x4 bb0 = *(const f32x4*)(b1 + cb),          bb1 = *(const f32x4*)(b1 + cb + 4);
#pragma unroll
      for (int e = 0; e < 4; e++) {
        float pr = fmaf(h[2], wc0[e], fmaf(h[1], wb0[e], fmaf(h[0], wa0[e], pb0[e])));
        float w  = kv0[e] - qv0[e] + pr;
        float t  = fmaf(w, gg0[e] * RSQ, bb0[e]);
        fa.v[r2 * 8 + e] = (__bf16)(t > 0.f ? t : 0.f);
        pr = fmaf(h[2], wc1[e], fmaf(h[1], wb1[e], fmaf(h[0], wa1[e], pb1[e])));
        w  = kv1[e] - qv1[e] + pr;
        t  = fmaf(w, gg1[e] * RSQ, bb1[e]);
        fa.v[r2 * 8 + 4 + e] = (__bf16)(t > 0.f ? t : 0.f);
      }
    }
#pragma unroll
    for (int nt = 0; nt < 4; nt++) {
      v16bf bf = ld_fragB(Wt1 + (size_t)(nt * 16 + s) * Cc, kb, half);
      acc[nt] = __builtin_amdgcn_wmma_f32_16x16x32_bf16(false, fa.v, false, bf, (short)0, acc[nt], false, false);
    }
  }
  // bias + bn2 + relu -> bf16 into LDS (t2, row = neighbor, col = hidden)
#pragma unroll
  for (int nt = 0; nt < 4; nt++) {
    int cc = nt * 16 + s;
    float bias = w1b[cc];
    float gv = g2[cc] * RSQ, bv = b2[cc];
#pragma unroll
    for (int i = 0; i < 8; i++) {
      float u = acc[nt][i] + bias;
      float t = fmaf(u, gv, bv);
      t2s[wv][i + 8 * half][cc] = f2bf(t > 0.f ? t : 0.f);
    }
  }
  __syncthreads();

  // ---- w2 GEMM: logits[16x64] = t2[16x64] @ Wt2^T
  v8f a2[4] = {{}, {}, {}, {}};
#pragma unroll
  for (int kb = 0; kb < 64; kb += 32) {
    v16bf fa2 = ld_fragA(&t2s[wv][s][0], kb, half);
#pragma unroll
    for (int nt = 0; nt < 4; nt++) {
      v16bf fb2 = ld_fragB(Wt2 + (size_t)(nt * 16 + s) * 64, kb, half);
      a2[nt] = __builtin_amdgcn_wmma_f32_16x16x32_bf16(false, fa2, false, fb2, (short)0, a2[nt], false, false);
    }
  }
#pragma unroll
  for (int nt = 0; nt < 4; nt++) {
    int cc = nt * 16 + s;
    float bias = w2b[cc];
#pragma unroll
    for (int i = 0; i < 8; i++) wts[wv][i + 8 * half][cc] = a2[nt][i] + bias;
  }
  __syncthreads();

  // ---- softmax over the 16 neighbors, per hidden column (2 cols per lane)
#pragma unroll
  for (int ci = 0; ci < 2; ci++) {
    int col = l + 32 * ci;
    float mx = -3.4e38f;
#pragma unroll
    for (int t = 0; t < 16; t++) mx = fmaxf(mx, wts[wv][t][col]);
    float sum = 0.f;
#pragma unroll
    for (int t = 0; t < 16; t++) { float e = __expf(wts[wv][t][col] - mx); wts[wv][t][col] = e; sum += e; }
    float inv = 1.f / sum;
#pragma unroll
    for (int t = 0; t < 16; t++) wts[wv][t][col] *= inv;
  }
  __syncthreads();

  // ---- out[c] = sum_s (xv[nbr_s][c] + pr[s][c]) * w[s][c%64]
  float* orow = out + (size_t)gid * Cc;
  for (int ch = 0; ch < 16; ch++) {
    int c = ch * 32 + l;
    int cc = c & 63;
    float wa = p2W[c], wb = p2W[Cc + c], wc = p2W[2 * Cc + c], pbv = p2b[c];
    float accv = 0.f;
#pragma unroll
    for (int t = 0; t < 16; t++) {
      int nb2 = nbs[wv][t];
      float pr = fmaf(hs[wv][t][2], wc, fmaf(hs[wv][t][1], wb, fmaf(hs[wv][t][0], wa, pbv)));
      float vv = xv[(size_t)(b * Np + nb2) * Cc + c];
      accv = fmaf(vv + pr, wts[wv][t][cc], accv);
    }
    orow[c] = accv;
  }
}

extern "C" void kernel_launch(void* const* d_in, const int* in_sizes, int n_in,
                              void* d_out, int out_size, void* d_ws, size_t ws_size,
                              hipStream_t stream) {
  const float* p   = (const float*)d_in[0];
  const float* x   = (const float*)d_in[1];
  const float* Wq  = (const float*)d_in[2];
  const float* bq  = (const float*)d_in[3];
  const float* Wk  = (const float*)d_in[4];
  const float* bk  = (const float*)d_in[5];
  const float* Wv  = (const float*)d_in[6];
  const float* bv  = (const float*)d_in[7];
  const float* p1W = (const float*)d_in[8];
  const float* p1b = (const float*)d_in[9];
  const float* pbg = (const float*)d_in[10];
  const float* pbb = (const float*)d_in[11];
  const float* p2W = (const float*)d_in[12];
  const float* p2b = (const float*)d_in[13];
  const float* g1  = (const float*)d_in[14];
  const float* b1  = (const float*)d_in[15];
  const float* w1W = (const float*)d_in[16];
  const float* w1b = (const float*)d_in[17];
  const float* g2  = (const float*)d_in[18];
  const float* b2  = (const float*)d_in[19];
  const float* w2W = (const float*)d_in[20];
  const float* w2b = (const float*)d_in[21];
  float* out = (float*)d_out;

  char* ws = (char*)d_ws;
  size_t off = 0;
  auto take = [&](size_t bytes) -> char* {
    char* r = ws + off; off += (bytes + 255) & ~(size_t)255; return r;
  };
  unsigned short* Abf = (unsigned short*)take(8192ull * 512 * 2); // x^T in bf16
  unsigned short* Wtq = (unsigned short*)take(512ull * 512 * 2);
  unsigned short* Wtk = (unsigned short*)take(512ull * 512 * 2);
  unsigned short* Wtv = (unsigned short*)take(512ull * 512 * 2);
  unsigned short* Wt1 = (unsigned short*)take(64ull * 512 * 2);
  unsigned short* Wt2 = (unsigned short*)take(64ull * 64 * 2);
  float* xqb = (float*)take(8192ull * 512 * 4);
  float* xkb = (float*)take(8192ull * 512 * 4);
  float* xvb = (float*)take(8192ull * 512 * 4);
  int*   idxb = (int*)take(8192ull * 16 * 4);

  knn_kernel<<<32, 256, 0, stream>>>(p, idxb);
  // x (B,512,512) -> A[b*512+n][cin] bf16 (per-batch transpose)
  tconv_kernel<<<dim3(16, 16, Bz), dim3(32, 8), 0, stream>>>(x, Abf, 512, 512, 512l * 512, 512l * 512);
  tconv_kernel<<<dim3(16, 16, 1), dim3(32, 8), 0, stream>>>(Wq, Wtq, 512, 512, 0, 0);
  tconv_kernel<<<dim3(16, 16, 1), dim3(32, 8), 0, stream>>>(Wk, Wtk, 512, 512, 0, 0);
  tconv_kernel<<<dim3(16, 16, 1), dim3(32, 8), 0, stream>>>(Wv, Wtv, 512, 512, 0, 0);
  tconv_kernel<<<dim3(2, 16, 1), dim3(32, 8), 0, stream>>>(w1W, Wt1, 512, 64, 0, 0);
  tconv_kernel<<<dim3(2, 2, 1), dim3(32, 8), 0, stream>>>(w2W, Wt2, 64, 64, 0, 0);

  gemm_kernel<<<dim3(4, 64), 256, 0, stream>>>(Abf, Wtq, bq, xqb);
  gemm_kernel<<<dim3(4, 64), 256, 0, stream>>>(Abf, Wtk, bk, xkb);
  gemm_kernel<<<dim3(4, 64), 256, 0, stream>>>(Abf, Wtv, bv, xvb);

  attn_kernel<<<2048, 128, 0, stream>>>(p, idxb, xqb, xkb, xvb, Wt1, Wt2,
                                        p1W, p1b, pbg, pbb, p2W, p2b,
                                        g1, b1, w1b, g2, b2, w2b, out);
}
